// Attention_12386685682407
// MI455X (gfx1250) — matrix operand order repaired and checked by reading, not compiled, not run on hardware
//
#include <hip/hip_runtime.h>
#include <hip/hip_bf16.h>

// Attention layer for MI455X (gfx1250), bf16 WMMA pipeline with TDM-staged GEMM.
// Workspace usage: ~184 MB of d_ws.

typedef __bf16 bf16_t;
typedef __attribute__((ext_vector_type(16))) __bf16 v16bf;
typedef __attribute__((ext_vector_type(8)))  __bf16 v8bf;
typedef __attribute__((ext_vector_type(8)))  float  v8f;
typedef __attribute__((ext_vector_type(4)))  unsigned int u32x4;
typedef __attribute__((ext_vector_type(8)))  int i32x8;
typedef __attribute__((ext_vector_type(4)))  int i32x4;

#define S_LEN   2048
#define D_MODEL 4096
#define NH      32
#define NKV     8
#define HDIM    128
#define HHD     (NH * HDIM)   // 4096
#define KVD     (NKV * HDIM)  // 1024

#if __has_builtin(__builtin_amdgcn_tensor_load_to_lds) && \
    __has_builtin(__builtin_amdgcn_s_wait_tensorcnt)
#define HAVE_TDM 1
#else
#define HAVE_TDM 0
#endif

// ---------------------------------------------------------------------------
// Fragment helpers (CDNA5 16x16x32 bf16 WMMA layouts, ISA 7.12.2)
// ---------------------------------------------------------------------------

__device__ __forceinline__ v8f zero8() {
  v8f z;
#pragma unroll
  for (int i = 0; i < 8; ++i) z[i] = 0.0f;
  return z;
}

__device__ __forceinline__ v16bf join8(v8bf lo, v8bf hi) {
  return __builtin_shufflevector(lo, hi, 0, 1, 2, 3, 4, 5, 6, 7,
                                 8, 9, 10, 11, 12, 13, 14, 15);
}

// A fragment (16 rows x 32 k) from row-major src (leading dim lda elements).
// Lane l: m = l&15, h = l>>4. elems 0..7 <- k = 8h.., elems 8..15 <- 16+8h..
__device__ __forceinline__ v16bf load_a_frag(const bf16_t* src, int lda,
                                             int lane) {
  int m = lane & 15, h = lane >> 4;
  const bf16_t* p = src + (size_t)m * lda;
  v8bf lo = *(const v8bf*)(p + 8 * h);
  v8bf hi = *(const v8bf*)(p + 16 + 8 * h);
  return join8(lo, hi);
}

// B fragment (32 k x 16 cols): column n of B = row n of row-major src (= B^T).
// Lane l: n = l&15, h = l>>4. elems 0..7 <- k = 8h.., elems 8..15 <- 16+8h..
__device__ __forceinline__ v16bf load_bt_frag(const bf16_t* src, int ldb,
                                              int lane) {
  int n = lane & 15, h = lane >> 4;
  const bf16_t* p = src + (size_t)n * ldb;
  v8bf lo = *(const v8bf*)(p + 8 * h);
  v8bf hi = *(const v8bf*)(p + 16 + 8 * h);
  return join8(lo, hi);
}

__device__ __forceinline__ v8f wmma_bf16(v16bf a, v16bf b, v8f c) {
  return __builtin_amdgcn_wmma_f32_16x16x32_bf16(false, a, false, b,
                                                 (short)0, c, false, false);
}

// ---------------------------------------------------------------------------
// Tensor Data Mover: 2D tile load (global -> LDS), bf16 elements.
// D# layout per CDNA5 ISA.
// lds_off: byte offset from LDS base (dynamic-shared assumed at offset 0).
// ---------------------------------------------------------------------------

#if HAVE_TDM
__device__ __forceinline__ void tdm_load_2d(unsigned lds_off, const void* gptr,
                                            unsigned tensor_d0,
                                            unsigned tensor_d1,
                                            unsigned tile_d0, unsigned tile_d1,
                                            unsigned stride0_elems) {
  unsigned long long ga = (unsigned long long)gptr;
  u32x4 g0;
  g0[0] = 1u;                                   // count=1, user descriptor
  g0[1] = lds_off;                              // lds_addr (bytes)
  g0[2] = (unsigned)(ga & 0xffffffffull);       // global_addr[31:0]
  g0[3] = (unsigned)((ga >> 32) & 0x1ffffffull) | (2u << 30);  // type=2
  i32x8 g1;
  g1[0] = (int)(1u << 16);                      // data_size=1 -> 2 bytes
  g1[1] = (int)((tensor_d0 & 0xffffu) << 16);   // tensor_dim0[15:0] @ [63:48]
  g1[2] = (int)(((tensor_d0 >> 16) & 0xffffu) | ((tensor_d1 & 0xffffu) << 16));
  g1[3] = (int)(((tensor_d1 >> 16) & 0xffffu) | (tile_d0 << 16));
  g1[4] = (int)(tile_d1 & 0xffffu);             // tile_dim1, tile_dim2=0 (2D)
  g1[5] = (int)stride0_elems;                   // tensor_dim0_stride[31:0]
  g1[6] = 0;                                    // stride hi / dim1_stride: 0
  g1[7] = 0;
  i32x4 gz = {0, 0, 0, 0};
#if defined(__clang_major__) && __clang_major__ >= 23
  i32x8 gz8 = {0, 0, 0, 0, 0, 0, 0, 0};
  __builtin_amdgcn_tensor_load_to_lds(g0, g1, gz, gz, gz8, 0);
#else
  __builtin_amdgcn_tensor_load_to_lds(g0, g1, gz, gz, 0);
#endif
}
#endif

// ---------------------------------------------------------------------------
// Elementwise / packing kernels
// ---------------------------------------------------------------------------

__global__ void k_cvt_bf16(const float* __restrict__ src,
                           bf16_t* __restrict__ dst, size_t n) {
  size_t i = (size_t)blockIdx.x * blockDim.x + threadIdx.x;
  if (i < n) dst[i] = (bf16_t)src[i];
}

// Transpose [K][N] f32 -> [N][K] bf16 (32x32 tiles via LDS).
__global__ void k_transpose_cvt(const float* __restrict__ src,
                                bf16_t* __restrict__ dst, int K, int N) {
  __shared__ float tile[32][33];
  int x  = blockIdx.x * 32 + threadIdx.x;
  int y0 = blockIdx.y * 32;
#pragma unroll
  for (int r = 0; r < 4; ++r)
    tile[threadIdx.y + 8 * r][threadIdx.x] =
        src[(size_t)(y0 + threadIdx.y + 8 * r) * N + x];
  __syncthreads();
#pragma unroll
  for (int r = 0; r < 4; ++r)
    dst[(size_t)(blockIdx.x * 32 + threadIdx.y + 8 * r) * K + y0 + threadIdx.x] =
        (bf16_t)tile[threadIdx.x][threadIdx.y + 8 * r];
}

// RoPE + pack: src f32 [S][Hh*128] -> dst bf16 [Hh][S][128]
__global__ void k_rope_pack(const float* __restrict__ src,
                            const float* __restrict__ cs,
                            const float* __restrict__ sn,
                            bf16_t* __restrict__ dst, int Hh) {
  unsigned idx = blockIdx.x * blockDim.x + threadIdx.x;
  unsigned total = (unsigned)S_LEN * Hh * 64;
  if (idx >= total) return;
  int i  = idx & 63;
  int hh = (idx >> 6) % Hh;
  int s  = idx / (64u * Hh);
  size_t sb = (size_t)s * (Hh * HDIM) + hh * HDIM + 2 * i;
  float t0 = src[sb], t1 = src[sb + 1];
  float c = cs[s * 64 + i], sv = sn[s * 64 + i];
  size_t ob = ((size_t)hh * S_LEN + s) * HDIM + 2 * i;
  dst[ob]     = (bf16_t)(t0 * c - t1 * sv);
  dst[ob + 1] = (bf16_t)(t0 * sv + t1 * c);
}

// Pack V transposed: src f32 [S][NKV*128] -> dst bf16 [NKV][128][S]
__global__ void k_pack_vt(const float* __restrict__ src,
                          bf16_t* __restrict__ dst) {
  unsigned idx = blockIdx.x * blockDim.x + threadIdx.x;
  if (idx >= (unsigned)S_LEN * NKV * HDIM) return;
  int s = idx & (S_LEN - 1);
  unsigned r = idx >> 11;
  int d  = r & (HDIM - 1);
  int hh = r >> 7;
  dst[((size_t)hh * HDIM + d) * S_LEN + s] =
      (bf16_t)src[(size_t)s * KVD + hh * HDIM + d];
}

// ---------------------------------------------------------------------------
// bf16 WMMA GEMM with TDM double-buffered LDS staging.
// C_f32[M][N] = A_bf16[M][K] @ B, B given as Bt_bf16[N][K].
// 256 threads = 8 waves; block tile 128(M) x 256(N); wave tile 64x64.
// Dynamic LDS: A[2][128][32] (16KB) ++ B[2][256][32] (32KB) = 48KB.
// ---------------------------------------------------------------------------

__global__ __launch_bounds__(256) void k_gemm_bf16(
    const bf16_t* __restrict__ A, const bf16_t* __restrict__ Bt,
    float* __restrict__ C, int M, int N, int K) {
  extern __shared__ __align__(16) bf16_t smem[];

  int tid = threadIdx.x, lane = tid & 31, wave = tid >> 5;
  int wm = wave >> 2, wn = wave & 3;  // 2 x 4 wave grid
  int rowblk = blockIdx.y * 128;
  int colblk = blockIdx.x * 256;
  int row0 = wm * 64;  // within LDS A tile
  int col0 = wn * 64;  // within LDS B tile
  bool issuer = (wave == 0);

  v8f acc[4][4];
#pragma unroll
  for (int i = 0; i < 4; ++i)
#pragma unroll
    for (int j = 0; j < 4; ++j) acc[i][j] = zero8();

  int nsteps = K / 32;
#if HAVE_TDM
  if (issuer) {
    tdm_load_2d(0u,     A  + (size_t)rowblk * K, K, M, 32, 128, K);
    tdm_load_2d(16384u, Bt + (size_t)colblk * K, K, N, 32, 256, K);
  }
#endif
  for (int s = 0; s < nsteps; ++s) {
    int b = s & 1;
    // buffer pointers / LDS byte offsets (plain runtime arithmetic; no
    // pointer arrays initialized from the extern-shared symbol)
    bf16_t* Ab = smem + b * 4096;           // A buf: 128x32 bf16
    bf16_t* Bb = smem + 8192 + b * 8192;    // B buf: 256x32 bf16
#if HAVE_TDM
    if (issuer) {
      if (s + 1 < nsteps) {
        int k0n = (s + 1) * 32;
        unsigned ldsAn = (unsigned)((b ^ 1) * 8192);
        unsigned ldsBn = 16384u + (unsigned)((b ^ 1) * 16384);
        tdm_load_2d(ldsAn, A  + (size_t)rowblk * K + k0n, K, M, 32, 128, K);
        tdm_load_2d(ldsBn, Bt + (size_t)colblk * K + k0n, K, N, 32, 256, K);
        __builtin_amdgcn_s_wait_tensorcnt(2);  // drain the 2 older loads
      } else {
        __builtin_amdgcn_s_wait_tensorcnt(0);
      }
    }
#else
    {  // cooperative fallback copy: global -> LDS
      int k0 = s * 32;
      for (int idx = tid; idx < 128 * 4; idx += 256) {
        int r = idx >> 2, seg = (idx & 3) * 8;
        *(v8bf*)(Ab + r * 32 + seg) =
            *(const v8bf*)(A + (size_t)(rowblk + r) * K + k0 + seg);
      }
      for (int idx = tid; idx < 256 * 4; idx += 256) {
        int r = idx >> 2, seg = (idx & 3) * 8;
        *(v8bf*)(Bb + r * 32 + seg) =
            *(const v8bf*)(Bt + (size_t)(colblk + r) * K + k0 + seg);
      }
    }
#endif
    __syncthreads();
    v16bf af[4], bfr[4];
#pragma unroll
    for (int i = 0; i < 4; ++i)
      af[i] = load_a_frag(Ab + (row0 + 16 * i) * 32, 32, lane);
#pragma unroll
    for (int j = 0; j < 4; ++j)
      bfr[j] = load_bt_frag(Bb + (col0 + 16 * j) * 32, 32, lane);
#pragma unroll
    for (int i = 0; i < 4; ++i)
#pragma unroll
      for (int j = 0; j < 4; ++j)
        acc[i][j] = wmma_bf16(af[i], bfr[j], acc[i][j]);
    __syncthreads();  // protect buffer b before it is refilled at step s+2
  }

  int nl = lane & 15, hl = lane >> 4;
#pragma unroll
  for (int i = 0; i < 4; ++i)
#pragma unroll
    for (int j = 0; j < 4; ++j)
#pragma unroll
      for (int v = 0; v < 8; ++v) {
        int r = rowblk + row0 + 16 * i + 8 * hl + v;
        int c = colblk + col0 + 16 * j + nl;
        C[(size_t)r * N + c] = acc[i][j][v];
      }
}

// ---------------------------------------------------------------------------
// Flash attention: one wave per (head, 16 query rows). Online softmax.
// Qb [H][S][128], Kb [NKV][S][128], Vt [NKV][128][S], Ob bf16 [S][H*128].
// ---------------------------------------------------------------------------

__global__ __launch_bounds__(256) void k_flash(
    const bf16_t* __restrict__ Qb, const bf16_t* __restrict__ Kb,
    const bf16_t* __restrict__ Vt, bf16_t* __restrict__ Ob) {
  __shared__ __align__(16) bf16_t pLds[8][16][32];  // per-wave P transpose
  int tid = threadIdx.x, lane = tid & 31, wlocal = tid >> 5;
  int wgid = blockIdx.x * 8 + wlocal;  // 0 .. H*(S/16)-1 = 4095
  int h   = wgid >> 7;
  int qt  = wgid & 127;
  int s0  = qt * 16;
  int hkv = h >> 2;  // GQA: n_rep = 4
  int nl = lane & 15, hl = lane >> 4;

  v16bf qf[4];
  const bf16_t* qbase = Qb + ((size_t)h * S_LEN + s0) * HDIM;
#pragma unroll
  for (int kk = 0; kk < 4; ++kk)
    qf[kk] = load_a_frag(qbase + 32 * kk, HDIM, lane);

  v8f o[8];
#pragma unroll
  for (int t = 0; t < 8; ++t) o[t] = zero8();
  float mrow[8], lrow[8];
#pragma unroll
  for (int j = 0; j < 8; ++j) { mrow[j] = -3.0e38f; lrow[j] = 0.0f; }

  const float scale = 0.08838834764831843f;  // 1/sqrt(128)
  int kend = s0 + 16;
  for (int jb = 0; jb < kend; jb += 32) {
    v8f st0 = zero8(), st1 = zero8();
    const bf16_t* k0p = Kb + ((size_t)hkv * S_LEN + jb) * HDIM;
    const bf16_t* k1p = k0p + 16 * HDIM;
    if (jb + 32 < kend)  // prefetch next key block (global_prefetch_b8)
      __builtin_prefetch(k0p + (size_t)32 * HDIM + (size_t)nl * HDIM, 0, 3);
#pragma unroll
    for (int kk = 0; kk < 4; ++kk) {
      v16bf b0 = load_bt_frag(k0p + 32 * kk, HDIM, lane);
      v16bf b1 = load_bt_frag(k1p + 32 * kk, HDIM, lane);
      st0 = wmma_bf16(qf[kk], b0, st0);
      st1 = wmma_bf16(qf[kk], b1, st1);
    }
    int key0 = jb + nl, key1 = jb + 16 + nl;
#pragma unroll
    for (int j = 0; j < 8; ++j) {
      int row = s0 + 8 * hl + j;
      float x0 = (key0 <= row) ? st0[j] * scale : -1.0e9f;
      float x1 = (key1 <= row) ? st1[j] * scale : -1.0e9f;
      float mx = fmaxf(x0, x1);
#pragma unroll
      for (int off = 1; off < 16; off <<= 1)
        mx = fmaxf(mx, __shfl_xor(mx, off, 32));
      float mnew  = fmaxf(mrow[j], mx);
      float alpha = __expf(mrow[j] - mnew);
      float p0 = __expf(x0 - mnew);
      float p1 = __expf(x1 - mnew);
      float rs = p0 + p1;
#pragma unroll
      for (int off = 1; off < 16; off <<= 1)
        rs += __shfl_xor(rs, off, 32);
      lrow[j] = lrow[j] * alpha + rs;
      mrow[j] = mnew;
#pragma unroll
      for (int t = 0; t < 8; ++t) o[t][j] *= alpha;
      pLds[wlocal][8 * hl + j][nl]      = (bf16_t)p0;
      pLds[wlocal][8 * hl + j][16 + nl] = (bf16_t)p1;
    }
    // wave-private LDS: DS-counter drain suffices, no workgroup barrier
    asm volatile("s_wait_dscnt 0x0" ::: "memory");
    v16bf pf = load_a_frag(&pLds[wlocal][0][0], 32, lane);
    const bf16_t* vtp = Vt + (size_t)hkv * HDIM * S_LEN + jb;
#pragma unroll
    for (int t = 0; t < 8; ++t) {
      v16bf vb = load_bt_frag(vtp + (size_t)(16 * t) * S_LEN, S_LEN, lane);
      o[t] = wmma_bf16(pf, vb, o[t]);
    }
  }
#pragma unroll
  for (int j = 0; j < 8; ++j) {
    float inv = 1.0f / lrow[j];
    int row = s0 + 8 * hl + j;
#pragma unroll
    for (int t = 0; t < 8; ++t)
      Ob[(size_t)row * HHD + h * HDIM + 16 * t + nl] = (bf16_t)(o[t][j] * inv);
  }
}

// ---------------------------------------------------------------------------
// Host launcher
// ---------------------------------------------------------------------------

extern "C" void kernel_launch(void* const* d_in, const int* in_sizes, int n_in,
                              void* d_out, int out_size, void* d_ws,
                              size_t ws_size, hipStream_t stream) {
  (void)in_sizes; (void)n_in; (void)out_size; (void)ws_size;
  const float* x  = (const float*)d_in[0];
  const float* wq = (const float*)d_in[1];
  const float* wk = (const float*)d_in[2];
  const float* wv = (const float*)d_in[3];
  const float* wo = (const float*)d_in[4];
  const float* fc = (const float*)d_in[5];
  const float* fs = (const float*)d_in[6];
  // d_in[7] = mask (causal; applied analytically), d_in[8..9] scalars unused.

  char* ws = (char*)d_ws;
  size_t off = 0;
  auto alloc = [&](size_t bytes) -> void* {
    void* p = ws + off;
    off += (bytes + 255) & ~(size_t)255;
    return p;
  };
  bf16_t* xb  = (bf16_t*)alloc((size_t)S_LEN * D_MODEL * 2);
  bf16_t* wqt = (bf16_t*)alloc((size_t)HHD * D_MODEL * 2);
  bf16_t* wkt = (bf16_t*)alloc((size_t)KVD * D_MODEL * 2);
  bf16_t* wvt = (bf16_t*)alloc((size_t)KVD * D_MODEL * 2);
  bf16_t* wot = (bf16_t*)alloc((size_t)D_MODEL * HHD * 2);
  float*  Qf  = (float*)alloc((size_t)S_LEN * HHD * 4);
  float*  Kf  = (float*)alloc((size_t)S_LEN * KVD * 4);
  float*  Vf  = (float*)alloc((size_t)S_LEN * KVD * 4);
  bf16_t* Qb  = (bf16_t*)alloc((size_t)NH  * S_LEN * HDIM * 2);
  bf16_t* Kb  = (bf16_t*)alloc((size_t)NKV * S_LEN * HDIM * 2);
  bf16_t* Vt  = (bf16_t*)alloc((size_t)NKV * HDIM * S_LEN * 2);
  bf16_t* Ob  = (bf16_t*)alloc((size_t)S_LEN * HHD * 2);

  const size_t gemm_lds = 49152;  // 48KB double-buffered A/B tiles

  {
    size_t n = (size_t)S_LEN * D_MODEL;
    k_cvt_bf16<<<(unsigned)(n / 256), 256, 0, stream>>>(x, xb, n);
  }
  {
    dim3 blk(32, 8);
    k_transpose_cvt<<<dim3(HHD / 32, D_MODEL / 32), blk, 0, stream>>>(wq, wqt, D_MODEL, HHD);
    k_transpose_cvt<<<dim3(KVD / 32, D_MODEL / 32), blk, 0, stream>>>(wk, wkt, D_MODEL, KVD);
    k_transpose_cvt<<<dim3(KVD / 32, D_MODEL / 32), blk, 0, stream>>>(wv, wvt, D_MODEL, KVD);
    k_transpose_cvt<<<dim3(D_MODEL / 32, HHD / 32), blk, 0, stream>>>(wo, wot, HHD, D_MODEL);
  }
  k_gemm_bf16<<<dim3(HHD / 256, S_LEN / 128), 256, gemm_lds, stream>>>(
      xb, wqt, Qf, S_LEN, HHD, D_MODEL);
  k_gemm_bf16<<<dim3(KVD / 256, S_LEN / 128), 256, gemm_lds, stream>>>(
      xb, wkt, Kf, S_LEN, KVD, D_MODEL);
  k_gemm_bf16<<<dim3(KVD / 256, S_LEN / 128), 256, gemm_lds, stream>>>(
      xb, wvt, Vf, S_LEN, KVD, D_MODEL);
  k_rope_pack<<<(S_LEN * NH  * 64) / 256, 256, 0, stream>>>(Qf, fc, fs, Qb, NH);
  k_rope_pack<<<(S_LEN * NKV * 64) / 256, 256, 0, stream>>>(Kf, fc, fs, Kb, NKV);
  k_pack_vt<<<(S_LEN * NKV * HDIM) / 256, 256, 0, stream>>>(Vf, Vt);
  k_flash<<<(NH * (S_LEN / 16)) / 8, 256, 0, stream>>>(Qb, Kb, Vt, Ob);
  k_gemm_bf16<<<dim3(D_MODEL / 256, S_LEN / 128), 256, gemm_lds, stream>>>(
      Ob, wot, (float*)d_out, S_LEN, D_MODEL, HHD);
}
